// TensorNeRF_4011499454666
// MI455X (gfx1250) — compile-verified, matrix operand order checked by reading, not run on hardware
//
#include <hip/hip_runtime.h>

typedef __attribute__((ext_vector_type(16))) _Float16 v16h;
typedef __attribute__((ext_vector_type(8)))  float    v8f;

#define NS 256
#define VGRID 160
#define AGRID 128
#define STEPSZ 0.0375f           // (3.0/160)*2
#define DISTV  (0.0375f*25.0f)   // step * dist_scale
#define KT1 13                   // ceil(390/32)
#define B1FRAGS (KT1*8*512)      // 53248 halves
#define B2FRAGS (4*8*512)        // 16384 halves
#define B3FRAGS (4*512)          // 2048 halves

__device__ __forceinline__ v8f wmma16(v16h a, v16h b, v8f c) {
  return __builtin_amdgcn_wmma_f32_16x16x32_f16(false, a, false, b, (short)0, c, false, false);
}

__device__ __forceinline__ float clampf(float v, float lo, float hi) {
  return fminf(fmaxf(v, lo), hi);
}

// trilinear sample, coords in [-1,1], align_corners semantics, volume [W][W][W]
__device__ __forceinline__ float trilin1(const float* __restrict__ vol, int W,
                                         float nx, float ny, float nz) {
  float fw = (float)(W - 1);
  float gx = clampf((nx + 1.f) * 0.5f * fw, 0.f, fw);
  float gy = clampf((ny + 1.f) * 0.5f * fw, 0.f, fw);
  float gz = clampf((nz + 1.f) * 0.5f * fw, 0.f, fw);
  int x0 = (int)gx, y0 = (int)gy, z0 = (int)gz;
  int x1 = min(x0 + 1, W - 1), y1 = min(y0 + 1, W - 1), z1 = min(z0 + 1, W - 1);
  float wx = gx - x0, wy = gy - y0, wz = gz - z0;
  size_t WW = (size_t)W * W;
  const float* p0 = vol + (size_t)z0 * WW;
  const float* p1 = vol + (size_t)z1 * WW;
  float v000 = p0[(size_t)y0 * W + x0], v001 = p0[(size_t)y0 * W + x1];
  float v010 = p0[(size_t)y1 * W + x0], v011 = p0[(size_t)y1 * W + x1];
  float v100 = p1[(size_t)y0 * W + x0], v101 = p1[(size_t)y0 * W + x1];
  float v110 = p1[(size_t)y1 * W + x0], v111 = p1[(size_t)y1 * W + x1];
  float c00 = v000 + (v001 - v000) * wx, c01 = v010 + (v011 - v010) * wx;
  float c10 = v100 + (v101 - v100) * wx, c11 = v110 + (v111 - v110) * wx;
  float c0 = c00 + (c01 - c00) * wy, c1 = c10 + (c11 - c10) * wy;
  return c0 + (c1 - c0) * wz;
}

// mlp_in[K] from F[0..29] = {feats[27], views[3]}
// layout: [feats(27), views(3), sin_f(162), cos_f(162), sin_v(18), cos_v(18)]
__device__ __forceinline__ float mlp_elem(const float* F, int K) {
  if (K >= 390) return 0.f;
  if (K < 30) return F[K];
  if (K < 354) {
    int j = K - 30;
    bool isCos = (j >= 162);
    if (isCos) j -= 162;
    float v = F[j / 6] * (float)(1 << (j % 6));
    return isCos ? __cosf(v) : __sinf(v);
  }
  int j = K - 354;
  bool isCos = (j >= 18);
  if (isCos) j -= 18;
  float v = F[27 + j / 6] * (float)(1 << (j % 6));
  return isCos ? __cosf(v) : __sinf(v);
}

// Pack w1/w2/w3 (row-major [K][N]) into per-lane WMMA B-fragment order:
// frag[((kt*NT + nt)*32 + lane)*16 + h], K = kt*32 + (h>>3)*16 + (lane>>4)*8 + (h&7)
__global__ void prep_frags(const float* __restrict__ w1, const float* __restrict__ w2,
                           const float* __restrict__ w3, _Float16* __restrict__ frag) {
  int idx = blockIdx.x * blockDim.x + threadIdx.x;
  int total = B1FRAGS + B2FRAGS + B3FRAGS;
  if (idx >= total) return;
  int e = idx;
  if (e < B1FRAGS) {
    int f = e / 512, rem = e % 512, lane = rem / 16, h = rem % 16;
    int kt = f / 8, nt = f % 8;
    int K = kt * 32 + ((h >> 3) << 4) + ((lane >> 4) << 3) + (h & 7);
    int N = nt * 16 + (lane & 15);
    frag[idx] = (_Float16)((K < 390) ? w1[K * 128 + N] : 0.f);
  } else if (e < B1FRAGS + B2FRAGS) {
    e -= B1FRAGS;
    int f = e / 512, rem = e % 512, lane = rem / 16, h = rem % 16;
    int kt = f / 8, nt = f % 8;
    int K = kt * 32 + ((h >> 3) << 4) + ((lane >> 4) << 3) + (h & 7);
    int N = nt * 16 + (lane & 15);
    frag[idx] = (_Float16)w2[K * 128 + N];
  } else {
    e -= (B1FRAGS + B2FRAGS);
    int kt = e / 512, rem = e % 512, lane = rem / 16, h = rem % 16;
    int K = kt * 32 + ((h >> 3) << 4) + ((lane >> 4) << 3) + (h & 7);
    int N = lane & 15;
    frag[idx] = (_Float16)((N < 3) ? w3[K * 3 + N] : 0.f);
  }
}

__global__ __launch_bounds__(256)
void nerf_fused(const float* __restrict__ rays_o, const float* __restrict__ rays_d,
                const float* __restrict__ density, const float* __restrict__ app,
                const float* __restrict__ alphaVol,
                const float* __restrict__ b1, const float* __restrict__ b2,
                const float* __restrict__ b3,
                const _Float16* __restrict__ frag,
                float* __restrict__ out, int nrays) {
  __shared__ float sNorm[NS][3];           // normalized sample coords
  __shared__ float sAW[NS];                // alpha, then weights (in-place scan)
  __shared__ float sFeats[8][16][32];      // per-wave feats+views
  __shared__ _Float16 sH[8][16][128];      // per-wave hidden activations (f16)
  __shared__ float rgbAcc[3];
  __shared__ float sAcc, sDepth;

  int ray = blockIdx.x;
  if (ray >= nrays) return;
  int tid = threadIdx.x;

  float ox = rays_o[ray * 3 + 0], oy = rays_o[ray * 3 + 1], oz = rays_o[ray * 3 + 2];
  float dx = rays_d[ray * 3 + 0], dy = rays_d[ray * 3 + 1], dz = rays_d[ray * 3 + 2];
  float vx = (dx == 0.f) ? 1e-6f : dx;
  float vy = (dy == 0.f) ? 1e-6f : dy;
  float vz = (dz == 0.f) ? 1e-6f : dz;
  float tx = fminf((1.5f - ox) / vx, (-1.5f - ox) / vx);
  float ty = fminf((1.5f - oy) / vy, (-1.5f - oy) / vy);
  float tz = fminf((1.5f - oz) / vz, (-1.5f - oz) / vz);
  float tmin = clampf(fmaxf(fmaxf(tx, ty), tz), 2.f, 6.f);

  if (tid < 3) rgbAcc[tid] = 0.f;

  // ---- phase 1: per-sample density/alpha ----
  {
    int s = tid;
    float t = tmin + STEPSZ * (float)s;
    float px = ox + dx * t, py = oy + dy * t, pz = oz + dz * t;
    bool inb = (px >= -1.5f) && (px <= 1.5f) && (py >= -1.5f) && (py <= 1.5f) &&
               (pz >= -1.5f) && (pz <= 1.5f);
    float nx = px * (1.f / 1.5f), ny = py * (1.f / 1.5f), nz = pz * (1.f / 1.5f);
    sNorm[s][0] = nx; sNorm[s][1] = ny; sNorm[s][2] = nz;
    float av = trilin1(alphaVol, AGRID, nx, ny, nz);
    float sf = trilin1(density, VGRID, nx, ny, nz);
    float sigma = 0.f;
    if (inb && av > 0.001f) {
      float x = sf - 10.f;
      sigma = (x > 20.f) ? x : log1pf(expf(x));  // softplus
    }
    sAW[s] = 1.f - expf(-sigma * DISTV);
  }
  __syncthreads();

  // ---- phase 2: serial transmittance scan ----
  if (tid == 0) {
    float T = 1.f, accw = 0.f, depth = 0.f;
    for (int i = 0; i < NS; ++i) {
      float a = sAW[i];
      float w = a * T;
      sAW[i] = w;
      accw += w;
      depth += w * (tmin + STEPSZ * (float)i);
      T *= (1.f - a + 1e-10f);
    }
    sAcc = accw;
    sDepth = depth + (1.f - accw) * (tmin + STEPSZ * (float)(NS - 1));
  }
  __syncthreads();

  // ---- phase 3: per-wave WMMA MLP over 16-sample tiles ----
  int wv = tid >> 5;
  int lane = tid & 31;
  int m = lane & 15;       // sample-in-tile (A rows / C cols)
  int g = lane >> 4;       // lane half
  const _Float16* b1f = frag;
  const _Float16* b2f = frag + B1FRAGS;
  const _Float16* b3f = frag + B1FRAGS + B2FRAGS;

  for (int tile = wv * 2; tile < wv * 2 + 2; ++tile) {
    int sBase = tile * 16;
    bool anyAct = __any(sAW[sBase + m] > 1e-4f) != 0;  // wave-uniform
    if (!anyAct) continue;

    // -- app-volume trilinear feats (27 ch) + views into LDS --
    {
      int sI = sBase + m;
      float nx = sNorm[sI][0], ny = sNorm[sI][1], nz = sNorm[sI][2];
      float fw = (float)(VGRID - 1);
      float gx = clampf((nx + 1.f) * 0.5f * fw, 0.f, fw);
      float gy = clampf((ny + 1.f) * 0.5f * fw, 0.f, fw);
      float gz = clampf((nz + 1.f) * 0.5f * fw, 0.f, fw);
      int x0 = (int)gx, y0 = (int)gy, z0 = (int)gz;
      int x1 = min(x0 + 1, VGRID - 1), y1 = min(y0 + 1, VGRID - 1), z1 = min(z0 + 1, VGRID - 1);
      float wx = gx - x0, wy = gy - y0, wz = gz - z0;
      float wx0 = 1.f - wx, wy0 = 1.f - wy, wz0 = 1.f - wz;
      size_t zs = (size_t)VGRID * VGRID;
      size_t i000 = (size_t)z0 * zs + (size_t)y0 * VGRID + x0;
      size_t i001 = (size_t)z0 * zs + (size_t)y0 * VGRID + x1;
      size_t i010 = (size_t)z0 * zs + (size_t)y1 * VGRID + x0;
      size_t i011 = (size_t)z0 * zs + (size_t)y1 * VGRID + x1;
      size_t i100 = (size_t)z1 * zs + (size_t)y0 * VGRID + x0;
      size_t i101 = (size_t)z1 * zs + (size_t)y0 * VGRID + x1;
      size_t i110 = (size_t)z1 * zs + (size_t)y1 * VGRID + x0;
      size_t i111 = (size_t)z1 * zs + (size_t)y1 * VGRID + x1;
      float w000 = wz0 * wy0 * wx0, w001 = wz0 * wy0 * wx;
      float w010 = wz0 * wy * wx0,  w011 = wz0 * wy * wx;
      float w100 = wz * wy0 * wx0,  w101 = wz * wy0 * wx;
      float w110 = wz * wy * wx0,   w111 = wz * wy * wx;
      int c0 = g ? 14 : 0, c1 = g ? 27 : 14;     // split 27 channels across lane halves
      for (int c = c0; c < c1; ++c) {
        const float* v = app + (size_t)c * ((size_t)VGRID * VGRID * VGRID);
        sFeats[wv][m][c] = v[i000] * w000 + v[i001] * w001 + v[i010] * w010 + v[i011] * w011 +
                           v[i100] * w100 + v[i101] * w101 + v[i110] * w110 + v[i111] * w111;
      }
      if (lane < 16) {
        sFeats[wv][m][27] = dx; sFeats[wv][m][28] = dy; sFeats[wv][m][29] = dz;
      }
    }
    __builtin_amdgcn_wave_barrier();
    const float* F = &sFeats[wv][m][0];

    // -- layer 1: [16x390(f16)] x [390x128] via 13 K-tiles x 8 N-tiles --
    v8f acc[8] = {};
    for (int kt = 0; kt < KT1; ++kt) {
      v16h a = {};
      #pragma unroll
      for (int h = 0; h < 16; ++h) {
        int K = kt * 32 + ((h >> 3) << 4) + (g << 3) + (h & 7);
        a[h] = (_Float16)mlp_elem(F, K);
      }
      #pragma unroll
      for (int nt = 0; nt < 8; ++nt) {
        v16h b = *((const v16h*)b1f + (kt * 8 + nt) * 32 + lane);
        acc[nt] = wmma16(a, b, acc[nt]);
      }
    }
    #pragma unroll
    for (int nt = 0; nt < 8; ++nt) {
      float bias = b1[nt * 16 + m];
      #pragma unroll
      for (int r = 0; r < 8; ++r) {
        float hv = fmaxf(acc[nt][r] + bias, 0.f);
        sH[wv][r + g * 8][nt * 16 + m] = (_Float16)hv;
      }
    }
    __builtin_amdgcn_wave_barrier();

    // -- layer 2: 128x128 --
    v8f acc2[8] = {};
    for (int kt = 0; kt < 4; ++kt) {
      v16h a = {};
      #pragma unroll
      for (int h = 0; h < 16; ++h) {
        int K = kt * 32 + ((h >> 3) << 4) + (g << 3) + (h & 7);
        a[h] = sH[wv][m][K];
      }
      #pragma unroll
      for (int nt = 0; nt < 8; ++nt) {
        v16h b = *((const v16h*)b2f + (kt * 8 + nt) * 32 + lane);
        acc2[nt] = wmma16(a, b, acc2[nt]);
      }
    }
    __builtin_amdgcn_wave_barrier();
    #pragma unroll
    for (int nt = 0; nt < 8; ++nt) {
      float bias = b2[nt * 16 + m];
      #pragma unroll
      for (int r = 0; r < 8; ++r) {
        float hv = fmaxf(acc2[nt][r] + bias, 0.f);
        sH[wv][r + g * 8][nt * 16 + m] = (_Float16)hv;
      }
    }
    __builtin_amdgcn_wave_barrier();

    // -- layer 3: 128x3 (padded to N=16) --
    v8f acc3 = {};
    for (int kt = 0; kt < 4; ++kt) {
      v16h a = {};
      #pragma unroll
      for (int h = 0; h < 16; ++h) {
        int K = kt * 32 + ((h >> 3) << 4) + (g << 3) + (h & 7);
        a[h] = sH[wv][m][K];
      }
      v16h b = *((const v16h*)b3f + kt * 32 + lane);
      acc3 = wmma16(a, b, acc3);
    }
    if (m < 3) {
      float bias = b3[m];
      #pragma unroll
      for (int r = 0; r < 8; ++r) {
        int mrow = r + g * 8;
        float wgt = sAW[sBase + mrow];
        float sig = 1.f / (1.f + __expf(-(acc3[r] + bias)));
        float contrib = (wgt > 1e-4f) ? wgt * sig : 0.f;
        atomicAdd(&rgbAcc[m], contrib);
      }
    }
  }
  __syncthreads();

  // ---- composite with white background ----
  if (tid < 3) out[ray * 3 + tid] = clampf(rgbAcc[tid] + (1.f - sAcc), 0.f, 1.f);
  if (tid == 3) out[nrays * 3 + ray] = sDepth;
}

extern "C" void kernel_launch(void* const* d_in, const int* in_sizes, int n_in,
                              void* d_out, int out_size, void* d_ws, size_t ws_size,
                              hipStream_t stream) {
  const float* rays_o   = (const float*)d_in[0];
  const float* rays_d   = (const float*)d_in[1];
  const float* density  = (const float*)d_in[2];
  const float* app      = (const float*)d_in[3];
  const float* alphaVol = (const float*)d_in[4];
  const float* w1 = (const float*)d_in[5];
  const float* b1 = (const float*)d_in[6];
  const float* w2 = (const float*)d_in[7];
  const float* b2 = (const float*)d_in[8];
  const float* w3 = (const float*)d_in[9];
  const float* b3 = (const float*)d_in[10];
  int nrays = in_sizes[0] / 3;

  _Float16* frag = (_Float16*)d_ws;
  int totalFrag = B1FRAGS + B2FRAGS + B3FRAGS;
  prep_frags<<<(totalFrag + 255) / 256, 256, 0, stream>>>(w1, w2, w3, frag);
  nerf_fused<<<nrays, 256, 0, stream>>>(rays_o, rays_d, density, app, alphaVol,
                                        b1, b2, b3, frag, (float*)d_out, nrays);
}
